// MultiHeadAttention_72541997630077
// MI455X (gfx1250) — matrix operand order repaired and checked by reading, not compiled, not run on hardware
//
#include <hip/hip_runtime.h>
#include <hip/hip_bf16.h>

typedef __attribute__((ext_vector_type(16))) __bf16 v16bf;
typedef __attribute__((ext_vector_type(8)))  __bf16 v8bf;
typedef __attribute__((ext_vector_type(8)))  float  v8f;

#define NB  8
#define NT  1024
#define NE  128
#define NH  8
#define NHE 1024
#define BT  (NB * NT)   /* 8192 */

// ---------------------------------------------------------------------------
// CDNA5 async global->LDS staging (ASYNCcnt-tracked, gfx1250)
// ---------------------------------------------------------------------------
static __device__ __forceinline__ void async_load_b128(unsigned lds_off,
                                                       const void* gaddr) {
  asm volatile("global_load_async_to_lds_b128 %0, %1, off"
               :: "v"(lds_off), "v"(gaddr) : "memory");
}
static __device__ __forceinline__ void wait_async0() {
  asm volatile("s_wait_asynccnt 0x0" ::: "memory");
}

// ---------------------------------------------------------------------------
// WMMA helpers (CDNA5 gfx1250, wave32)
// ---------------------------------------------------------------------------
static __device__ __forceinline__ v8f wmma_bf16(v16bf a, v16bf b, v8f c) {
  return __builtin_amdgcn_wmma_f32_16x16x32_bf16(false, a, false, b, (short)0, c,
                                                 false, false);
}

static __device__ __forceinline__ v8f vzero8() {
  v8f z;
#pragma unroll
  for (int i = 0; i < 8; ++i) z[i] = 0.0f;
  return z;
}

// A-operand fragment (16x32 bf16): lane%16 picks row; elems 0..7 <- K=k0+8*(lane/16)+i,
// elems 8..15 <- +16.  (ISA 7.12.2)
static __device__ __forceinline__ v16bf load_a_frag(const __bf16* __restrict__ base,
                                                    int stride, int row0, int k0) {
  const int lane = threadIdx.x & 31;
  const int r    = row0 + (lane & 15);
  const int hi   = lane >> 4;
  const __bf16* p = base + (size_t)r * stride + k0 + hi * 8;
  v8bf lo = *(const v8bf*)(p);
  v8bf up = *(const v8bf*)(p + 16);
  v16bf f;
#pragma unroll
  for (int i = 0; i < 8; ++i) { f[i] = lo[i]; f[i + 8] = up[i]; }
  return f;
}

// B-operand fragment for B = W^T, W row-major [N,K]: lane holds col N=n0+lane%16,
// elems 0..7 <- K = k0 + 8*(lane/16)+e, elems 8..15 <- +16  (two 16B loads).
static __device__ __forceinline__ v16bf load_bt_frag(const __bf16* __restrict__ base,
                                                     int stride, int n0, int k0) {
  const int lane = threadIdx.x & 31;
  const int r    = n0 + (lane & 15);
  const int hi   = lane >> 4;
  const __bf16* p = base + (size_t)r * stride + k0 + hi * 8;
  v8bf lo = *(const v8bf*)(p);
  v8bf up = *(const v8bf*)(p + 16);
  v16bf f;
#pragma unroll
  for (int i = 0; i < 8; ++i) { f[i] = lo[i]; f[i + 8] = up[i]; }
  return f;
}

// Fragment loaders reading from an LDS-resident row-major tile
static __device__ __forceinline__ v16bf lds_a_frag(const __bf16* lds, int stride,
                                                   int row0) {
  const int lane = threadIdx.x & 31;
  const int r    = row0 + (lane & 15);
  const int hi   = lane >> 4;
  const __bf16* p = lds + r * stride + hi * 8;
  v8bf lo = *(const v8bf*)(p);
  v8bf up = *(const v8bf*)(p + 16);
  v16bf f;
#pragma unroll
  for (int i = 0; i < 8; ++i) { f[i] = lo[i]; f[i + 8] = up[i]; }
  return f;
}
static __device__ __forceinline__ v16bf lds_bt_frag(const __bf16* lds, int stride,
                                                    int n0, int k0) {
  const int lane = threadIdx.x & 31;
  const int r    = n0 + (lane & 15);
  const int hi   = lane >> 4;
  const __bf16* p = lds + r * stride + k0 + hi * 8;
  v8bf lo = *(const v8bf*)(p);
  v8bf up = *(const v8bf*)(p + 16);
  v16bf f;
#pragma unroll
  for (int i = 0; i < 8; ++i) { f[i] = lo[i]; f[i + 8] = up[i]; }
  return f;
}

// ---------------------------------------------------------------------------
// Kernel 0: f32 -> bf16 convert with scale
// ---------------------------------------------------------------------------
__global__ void cvt_f32_bf16_kernel(const float* __restrict__ src,
                                    __bf16* __restrict__ dst, int n, float scale) {
  int i = blockIdx.x * blockDim.x + threadIdx.x;
  if (i < n) dst[i] = (__bf16)(src[i] * scale);
}

// ---------------------------------------------------------------------------
// Kernel 1: projection GEMM  out[M,NHE] = X[M,NE] @ W^T   (W is [NHE,NE])
// Block = 8 waves; tile 128(M) x 64(N). Shared W tile (64x128 = 16KB) staged
// into LDS once per block via async DMA; all waves read B-frags from LDS.
// TRANSPOSED=1 stores V as Vt[b][h][e][t] with packed 16B stores.
// ---------------------------------------------------------------------------
template <int TRANSPOSED>
__global__ void __launch_bounds__(256)
proj_kernel(const __bf16* __restrict__ X, const __bf16* __restrict__ W,
            __bf16* __restrict__ out) {
  __shared__ __bf16 ldsW[64 * NE];   // 16 KB

  const int lane = threadIdx.x & 31;
  const int wave = threadIdx.x >> 5;
  const int l16  = lane & 15;
  const int hi   = lane >> 4;
  const int m0   = blockIdx.x * 128 + wave * 16;
  const int nb   = blockIdx.y * 64;

  // ---- async stage W[nb..nb+63][0..127] into LDS (4 x b128 per thread) ----
#pragma unroll
  for (int j = 0; j < 4; ++j) {
    const int idx  = threadIdx.x + j * 256;      // [0,1024): 16B chunks
    const int row  = idx >> 4;                   // 16 chunks per 128-elem row
    const int part = idx & 15;
    async_load_b128((unsigned)((row * NE + part * 8) * 2),
                    W + (size_t)(nb + row) * NE + part * 8);
  }

  v16bf af[4];
#pragma unroll
  for (int j = 0; j < 4; ++j) af[j] = load_a_frag(X, NE, m0, j * 32);

  wait_async0();
  __syncthreads();

#pragma unroll
  for (int nt = 0; nt < 4; ++nt) {
    const int n0 = nb + nt * 16;
    v8f acc = vzero8();
#pragma unroll
    for (int j = 0; j < 4; ++j)
      acc = wmma_bf16(af[j], lds_bt_frag(ldsW, NE, nt * 16, j * 32), acc);

    const int n = n0 + l16;
    if (TRANSPOSED) {
      // rows m are consecutive over v -> one packed 16B store into Vt[b][h][e][t]
      const int mb = m0 + 8 * hi;
      const int b  = mb >> 10, t0 = mb & (NT - 1);
      const int h  = n >> 7,   ec = n & (NE - 1);
      v8bf pk;
#pragma unroll
      for (int v = 0; v < 8; ++v) pk[v] = (__bf16)acc[v];
      *(v8bf*)(out + ((size_t)(b * NH + h) * NE + ec) * NT + t0) = pk;
    } else {
#pragma unroll
      for (int v = 0; v < 8; ++v)
        out[(size_t)(m0 + v + 8 * hi) * NHE + n] = (__bf16)acc[v];
    }
  }
}

// ---------------------------------------------------------------------------
// Kernel 2: flash attention per (b, h, 128-row q block). 8 waves, one 16-row
// q tile each. Scores computed TRANSPOSED (St = K_tile x Q^T) so the St C/D
// layout feeds the P.V WMMA A-operand directly (no LDS, no cross-lane moves).
// Scores arrive pre-multiplied by log2(e) (folded into Wq/Wk), so softmax
// runs in the base-2 domain with raw v_exp_f32 (exp2).
// ---------------------------------------------------------------------------
__global__ void __launch_bounds__(256)
attn_kernel(const __bf16* __restrict__ Qb, const __bf16* __restrict__ Kb,
            const __bf16* __restrict__ Vt, const float* __restrict__ mask_q,
            const float* __restrict__ mask_k, __bf16* __restrict__ Ab) {
  const int lane = threadIdx.x & 31;
  const int wave = threadIdx.x >> 5;
  const int l16  = lane & 15;
  const int hi   = lane >> 4;
  const int b    = blockIdx.z;
  const int h    = blockIdx.y;
  const int qblk = blockIdx.x * 128;
  const int q0   = qblk + wave * 16;

  const __bf16* Qh = Qb + (size_t)b * NT * NHE + h * NE;  // row stride NHE
  const __bf16* Kh = Kb + (size_t)b * NT * NHE + h * NE;  // row stride NHE
  const __bf16* Vh = Vt + (size_t)(b * NH + h) * NE * NT; // row stride NT

  v16bf qf[4];
#pragma unroll
  for (int j = 0; j < 4; ++j) qf[j] = load_bt_frag(Qh, NHE, q0, j * 32);

  const int   qidx = q0 + l16;               // this lane's q row (St column)
  const float mq   = mask_q[b * NT + qidx];

  float mmax = -__builtin_inff();
  float ssum = 0.0f;
  v8f o[8];
#pragma unroll
  for (int nt = 0; nt < 8; ++nt) o[nt] = vzero8();

  int kend = ((q0 + 16 + 31) >> 5) << 5;     // per-wave causal bound
  if (kend > NT) kend = NT;

  for (int kt = 0; kt < kend; kt += 32) {
    // ---- transposed scores: St[k, q] for k in [kt, kt+32) ----
    v8f st[2];
#pragma unroll
    for (int t2 = 0; t2 < 2; ++t2) {
      v8f s = vzero8();
#pragma unroll
      for (int j = 0; j < 4; ++j)
        s = wmma_bf16(load_a_frag(Kh, NHE, kt + t2 * 16, j * 32), qf[j], s);
      // masks: causal (-inf), then padding override (-1e10) -- reference order
      v8f mk = *(const v8f*)(mask_k + (size_t)b * NT + kt + t2 * 16 + 8 * hi);
#pragma unroll
      for (int v = 0; v < 8; ++v) {
        const int kidx = kt + t2 * 16 + v + 8 * hi;
        float sv = s[v];
        sv = (kidx > qidx) ? -__builtin_inff() : sv;
        sv = (mq == 0.0f || mk[v] == 0.0f) ? -1.0e10f : sv;
        s[v] = sv;
      }
      st[t2] = s;
    }

    // ---- online softmax stats (base-2 domain, per q row = lane's column) ----
    float cm = fmaxf(st[0][0], st[1][0]);
#pragma unroll
    for (int v = 1; v < 8; ++v) cm = fmaxf(cm, fmaxf(st[0][v], st[1][v]));
    cm = fmaxf(cm, __shfl_xor(cm, 16));
    const float nm   = fmaxf(mmax, cm);
    const float corr = __builtin_amdgcn_exp2f(mmax - nm);

    // ---- P fragment: St layout IS the A-operand layout for P.V ----
    v16bf pA;
    float ps = 0.0f;
#pragma unroll
    for (int v = 0; v < 8; ++v) {
      const float p0 = __builtin_amdgcn_exp2f(st[0][v] - nm);
      const float p1 = __builtin_amdgcn_exp2f(st[1][v] - nm);
      pA[v]     = (__bf16)p0;
      pA[v + 8] = (__bf16)p1;
      ps += p0 + p1;
    }
    ps += __shfl_xor(ps, 16);
    ssum = ssum * corr + ps;
    mmax = nm;

    float f[8];
#pragma unroll
    for (int v = 0; v < 8; ++v) f[v] = __shfl(corr, v + 8 * hi);

#pragma unroll
    for (int nt = 0; nt < 8; ++nt) {
#pragma unroll
      for (int v = 0; v < 8; ++v) o[nt][v] *= f[v];
      o[nt] = wmma_bf16(pA, load_bt_frag(Vh, NT, nt * 16, kt), o[nt]);
    }
  }

  // ---- normalize and store attention output (bf16, [b][t][h*NE+e]) ----
  const float inv = 1.0f / ssum;
  float fi[8];
#pragma unroll
  for (int v = 0; v < 8; ++v) fi[v] = __shfl(inv, v + 8 * hi);

#pragma unroll
  for (int nt = 0; nt < 8; ++nt) {
    const int col = h * NE + nt * 16 + l16;
#pragma unroll
    for (int v = 0; v < 8; ++v) {
      const int t = q0 + v + 8 * hi;
      Ab[((size_t)b * NT + t) * NHE + col] = (__bf16)(o[nt][v] * fi[v]);
    }
  }
}

// ---------------------------------------------------------------------------
// Kernel 3: output projection, computed TRANSPOSED: D = Wu x Ab^T.
//   A-operand = Wu rows [n=128, K=1024] (staged per-chunk in LDS, 8KB)
//   B-operand = Ab rows [m,     K=1024] (two 16B loads)
// Output columns n are consecutive over the accumulator index -> packed
// 32B f32 epilogue. Block = 8 waves; wave = 16(m) x 128(n).
// ---------------------------------------------------------------------------
__global__ void __launch_bounds__(256)
outproj_kernel(const __bf16* __restrict__ Ab, const __bf16* __restrict__ Wub,
               const float* __restrict__ bu, float* __restrict__ Y) {
  __shared__ __bf16 ldsW[NE * 32];   // 8 KB: Wu[:,kk:kk+32]

  const int lane = threadIdx.x & 31;
  const int wave = threadIdx.x >> 5;
  const int l16  = lane & 15;
  const int hi   = lane >> 4;
  const int m0   = blockIdx.x * 128 + wave * 16;

  v8f acc[8];
#pragma unroll
  for (int na = 0; na < 8; ++na) acc[na] = vzero8();

  for (int kk = 0; kk < NHE; kk += 32) {
    __syncthreads();   // previous chunk's LDS reads complete before overwrite
#pragma unroll
    for (int j = 0; j < 2; ++j) {
      const int idx  = threadIdx.x + j * 256;   // [0,512): 16B chunks
      const int row  = idx >> 2;                // 4 chunks per 32-elem row
      const int part = idx & 3;
      async_load_b128((unsigned)((row * 32 + part * 8) * 2),
                      Wub + (size_t)row * NHE + kk + part * 8);
    }
    if (kk + 32 < NHE)
      __builtin_prefetch(Ab + (size_t)(m0 + l16) * NHE + kk + 32, 0, 3);

    v16bf bfr = load_bt_frag(Ab, NHE, m0, kk);  // B = Ab^T fragment (global)

    wait_async0();
    __syncthreads();

#pragma unroll
    for (int na = 0; na < 8; ++na)
      acc[na] = wmma_bf16(lds_a_frag(ldsW, 32, na * 16), bfr, acc[na]);
  }

  // ---- epilogue: Y[m][n] = acc + bu[n], packed 32B loads/stores ----
  const int m = m0 + l16;
#pragma unroll
  for (int na = 0; na < 8; ++na) {
    const int nbase = na * 16 + 8 * hi;
    v8f bb = *(const v8f*)(bu + nbase);
    v8f r;
#pragma unroll
    for (int v = 0; v < 8; ++v) r[v] = acc[na][v] + bb[v];
    *(v8f*)(Y + (size_t)m * NE + nbase) = r;
  }
}

// ---------------------------------------------------------------------------
// Host launcher
// ---------------------------------------------------------------------------
extern "C" void kernel_launch(void* const* d_in, const int* in_sizes, int n_in,
                              void* d_out, int out_size, void* d_ws, size_t ws_size,
                              hipStream_t stream) {
  const float* q      = (const float*)d_in[0];
  const float* k      = (const float*)d_in[1];
  const float* mask_q = (const float*)d_in[2];
  const float* mask_k = (const float*)d_in[3];
  const float* Wq     = (const float*)d_in[4];
  const float* Wk     = (const float*)d_in[5];
  const float* Wv     = (const float*)d_in[6];
  const float* Wu     = (const float*)d_in[7];
  const float* bu     = (const float*)d_in[8];
  float* Y            = (float*)d_out;

  __bf16* w   = (__bf16*)d_ws;
  __bf16* qb  = w;                               // BT*NE
  __bf16* kbi = qb  + (size_t)BT * NE;           // BT*NE
  __bf16* Wqb = kbi + (size_t)BT * NE;           // NHE*NE
  __bf16* Wkb = Wqb + (size_t)NHE * NE;
  __bf16* Wvb = Wkb + (size_t)NHE * NE;
  __bf16* Wub = Wvb + (size_t)NHE * NE;          // NE*NHE
  __bf16* Qp  = Wub + (size_t)NHE * NE;          // BT*NHE
  __bf16* Kp  = Qp  + (size_t)BT * NHE;
  __bf16* Vtp = Kp  + (size_t)BT * NHE;          // [b][h][e][t]
  __bf16* Abp = Vtp + (size_t)BT * NHE;

  // 128^(-1/4) * sqrt(log2(e)): score scale AND base-2 softmax fold, split
  // evenly between Wq and Wk.
  const float scale = 0.29730177875068026f * 1.2011224087864498f;

  {
    const int nAct = BT * NE;
    const int nW   = NHE * NE;
    cvt_f32_bf16_kernel<<<(nAct + 255) / 256, 256, 0, stream>>>(q,  qb,  nAct, 1.0f);
    cvt_f32_bf16_kernel<<<(nAct + 255) / 256, 256, 0, stream>>>(k,  kbi, nAct, 1.0f);
    cvt_f32_bf16_kernel<<<(nW   + 255) / 256, 256, 0, stream>>>(Wq, Wqb, nW, scale);
    cvt_f32_bf16_kernel<<<(nW   + 255) / 256, 256, 0, stream>>>(Wk, Wkb, nW, scale);
    cvt_f32_bf16_kernel<<<(nW   + 255) / 256, 256, 0, stream>>>(Wv, Wvb, nW, 1.0f);
    cvt_f32_bf16_kernel<<<(nW   + 255) / 256, 256, 0, stream>>>(Wu, Wub, nW, 1.0f);
  }

  {
    dim3 g(BT / 128, NHE / 64);
    proj_kernel<0><<<g, 256, 0, stream>>>(qb,  Wqb, Qp);
    proj_kernel<0><<<g, 256, 0, stream>>>(kbi, Wkb, Kp);
    proj_kernel<1><<<g, 256, 0, stream>>>(kbi, Wvb, Vtp);   // V stored transposed
  }

  {
    dim3 g(NT / 128, NH, NB);
    attn_kernel<<<g, 256, 0, stream>>>(Qp, Kp, Vtp, mask_q, mask_k, Abp);
  }

  outproj_kernel<<<BT / 128, 256, 0, stream>>>(Abp, Wub, bu, Y);
}
